// CrossAttUnit_80436147519809
// MI455X (gfx1250) — matrix operand order repaired and checked by reading, not compiled, not run on hardware
//
#include <hip/hip_runtime.h>
#include <hip/hip_bf16.h>

// ---------------------------------------------------------------------------
// Problem constants (fixed by setup_inputs)
// ---------------------------------------------------------------------------
#define NROWS   262144
#define DDIM    256
#define HDIM    256
#define ADIM    64
#define SEGLEN  128
#define NSEG    (NROWS / SEGLEN)     // 2048
#define EPSV    1e-6f
#define INVL    (1.0f / 128.0f)

// f16 weight cache layout inside workspace (element offsets)
#define W0K_OFF 0
#define W1K_OFF 65536
#define W0Q_OFF 81920
#define W1Q_OFF 147456
#define WTOTAL  163840

typedef __attribute__((ext_vector_type(16))) _Float16 v16h;
typedef __attribute__((ext_vector_type(8)))  _Float16 v8h;
typedef __attribute__((ext_vector_type(8)))  float    v8f;

static __device__ inline v16h cat8(v8h lo, v8h hi) {
  return __builtin_shufflevector(lo, hi, 0,1,2,3,4,5,6,7,8,9,10,11,12,13,14,15);
}

// WMMA A fragment (16x32, f16) built from an fp32 row-major matrix.
// ISA layout: lane (half = lane>>4, m = lane&15) holds row m; elements 0..7
// are K = col0 + 8*half + e, elements 8..15 are K = col0 + 16 + 8*half + e.
static __device__ inline v16h a_frag_f32(const float* __restrict__ base, int ld,
                                         int row, int col0, int half) {
  const float* p = base + (size_t)row * ld + col0 + 8 * half;
  v16h a;
#pragma unroll
  for (int e = 0; e < 8; ++e) { a[e] = (_Float16)p[e]; a[e + 8] = (_Float16)p[e + 16]; }
  return a;
}

// A fragment from native f16 storage (global or LDS).
static __device__ inline v16h a_frag_f16(const _Float16* base, int ld,
                                         int row, int col0, int half) {
  const _Float16* p = base + (size_t)row * ld + col0 + 8 * half;
  v8h lo = *(const v8h*)p;          // 16B aligned
  v8h hi = *(const v8h*)(p + 16);
  return cat8(lo, hi);
}

// WMMA B fragment (32x16, f16) from f16 row-major W[nOut][K] (implicit W^T).
// Lane n holds column N = n&15; elements 0..7 are K = col0 + 8*(n>>4) + e,
// elements 8..15 are K = col0 + 16 + 8*(n>>4) + e.
static __device__ inline v16h b_frag_f16(const _Float16* base, int ld,
                                         int nrow, int col0, int half) {
  const _Float16* p = base + (size_t)nrow * ld + col0 + 8 * half; // 16B aligned
  return cat8(*(const v8h*)p, *(const v8h*)(p + 16));
}

static __device__ inline v8f wmma32(v16h a, v16h b, v8f c) {
  // D = A(16x32) * B(32x16) + C, fp32 accumulate
  return __builtin_amdgcn_wmma_f32_16x16x32_f16(false, a, false, b,
                                                (short)0, c, false, false);
}

// ---------------------------------------------------------------------------
// Phase 0: one-shot fp32 -> f16 conversion of all four weight matrices.
// ---------------------------------------------------------------------------
__global__ __launch_bounds__(256)
void cvt_weights_kernel(const float* __restrict__ w0k, const float* __restrict__ w1k,
                        const float* __restrict__ w0q, const float* __restrict__ w1q,
                        _Float16* __restrict__ dst) {
  const int idx = blockIdx.x * 256 + threadIdx.x;     // 0 .. WTOTAL-1
  if (idx >= WTOTAL) return;
  const float* src; int off;
  if (idx < W1K_OFF)      { src = w0k; off = W0K_OFF; }
  else if (idx < W0Q_OFF) { src = w1k; off = W1K_OFF; }
  else if (idx < W1Q_OFF) { src = w0q; off = W0Q_OFF; }
  else                    { src = w1q; off = W1Q_OFF; }
  dst[idx] = (_Float16)src[idx - off];
}

// ---------------------------------------------------------------------------
// Phase 1: fused 2-layer MLP  out = (leaky(X @ W0^T + b0)) @ W1^T + b1
// One workgroup = 128 rows of X. W0/W1 (f16) are staged once into LDS and
// shared by all 8 waves (8x less L2 traffic); the hidden [128,256] tile also
// lives in LDS. Padded stride 264 halves -> conflict-free strided fragment
// gathers. Total LDS = 231 KB of the WGP's 320 KB (1 WG / WGP, 8 waves).
// Biases are hoisted into registers so the WMMA loop has no global loads.
// ---------------------------------------------------------------------------
__global__ __launch_bounds__(256)
void proj_kernel(const float* __restrict__ X,        // [NROWS, 256] fp32
                 const _Float16* __restrict__ W0,    // [256, 256] f16
                 const float* __restrict__ b0,       // [256]
                 const _Float16* __restrict__ W1,    // [64, 256] f16
                 const float* __restrict__ b1,       // [64]
                 _Float16* __restrict__ outp)        // [NROWS, 64] f16
{
  __shared__ _Float16 w0sh[256][264];
  __shared__ _Float16 w1sh[64][264];
  __shared__ _Float16 hsh[128][264];

  const int tid  = threadIdx.x;
  const int tile = blockIdx.x;          // 0..2047, 128 rows each
  const int lane = tid & 31;
  const int wv   = tid >> 5;            // 0..7 (wave32)
  const int half = lane >> 4;
  const int lrow = lane & 15;
  const int row  = tile * 128 + wv * 16 + lrow;   // this lane's A row

  // Cooperative stage of W0 (8192 x 16B) and W1 (2048 x 16B) into LDS.
  for (int c = tid; c < 256 * 32; c += 256) {
    const int r = c >> 5, cc = (c & 31) * 8;
    *(v8h*)&w0sh[r][cc] = *(const v8h*)(W0 + (size_t)r * HDIM + cc);
  }
  for (int c = tid; c < 64 * 32; c += 256) {
    const int r = c >> 5, cc = (c & 31) * 8;
    *(v8h*)&w1sh[r][cc] = *(const v8h*)(W1 + (size_t)r * HDIM + cc);
  }

  __builtin_prefetch(X + (size_t)(row + 16) * DDIM, 0, 0);   // global_prefetch_b8

  // Hoist all per-lane bias values out of the WMMA loops.
  float bias0[16], bias1[4];
#pragma unroll
  for (int c = 0; c < 16; ++c) bias0[c] = b0[c * 16 + lrow];
#pragma unroll
  for (int c = 0; c < 4; ++c)  bias1[c] = b1[c * 16 + lrow];

  // Overlap with LDS staging: preload this wave's 16x256 A strip as 8
  // K-step fragments (fp32 global -> f16 registers).
  v16h afrag[8];
#pragma unroll
  for (int kk = 0; kk < 8; ++kk)
    afrag[kk] = a_frag_f32(X, DDIM, row, kk * 32, half);

  __syncthreads();

  // Layer 1: 16 column tiles of 16, K = 256 (8 WMMA steps each), B from LDS.
  for (int c = 0; c < 16; ++c) {
    v8f acc = {};
#pragma unroll
    for (int kk = 0; kk < 8; ++kk) {
      v16h bf = b_frag_f16(&w0sh[0][0], 264, c * 16 + lrow, kk * 32, half);
      acc = wmma32(afrag[kk], bf, acc);
    }
    const int col = c * 16 + lrow;
    const float bias = bias0[c];
#pragma unroll
    for (int j = 0; j < 8; ++j) {
      float v = acc[j] + bias;
      v = v > 0.0f ? v : 0.01f * v;                  // leaky_relu
      hsh[wv * 16 + j + 8 * half][col] = (_Float16)v;
    }
  }
  __syncthreads();

  // Layer 2: A from LDS hidden tile, B from LDS W1, 4 col tiles (ADIM = 64).
  v16h h2[8];
#pragma unroll
  for (int kk = 0; kk < 8; ++kk)
    h2[kk] = a_frag_f16(&hsh[0][0], 264, wv * 16 + lrow, kk * 32, half);

#pragma unroll
  for (int c = 0; c < 4; ++c) {
    v8f acc = {};
#pragma unroll
    for (int kk = 0; kk < 8; ++kk) {
      v16h bf = b_frag_f16(&w1sh[0][0], 264, c * 16 + lrow, kk * 32, half);
      acc = wmma32(h2[kk], bf, acc);
    }
    const int col = c * 16 + lrow;
    const float bias = bias1[c];
#pragma unroll
    for (int j = 0; j < 8; ++j) {
      float v = acc[j] + bias;
      outp[(size_t)(tile * 128 + wv * 16 + j + 8 * half) * ADIM + col] = (_Float16)v;
    }
  }
}

// ---------------------------------------------------------------------------
// Phase 2: per-segment attention + softmax + loss + 5 Sinkhorn iterations.
// One workgroup = one 128x128 segment. The q segment is staged once in LDS
// (stride 72 -> conflict-free B gathers); all iterative work stays in LDS.
// ---------------------------------------------------------------------------
__global__ __launch_bounds__(256)
void attn_kernel(const _Float16* __restrict__ kp,  // [NROWS, 64]
                 const _Float16* __restrict__ qp,  // [NROWS, 64]
                 float* __restrict__ Wout,         // [NSEG*128*128]
                 float* __restrict__ loss)         // scalar accumulator
{
  __shared__ float Msh[128][129];
  __shared__ float red[128];
  __shared__ _Float16 qsh[128][72];

  const int seg  = blockIdx.x;
  const int tid  = threadIdx.x;
  const int lane = tid & 31;
  const int wv   = tid >> 5;
  const int half = lane >> 4;
  const int lrow = lane & 15;
  const int rb   = seg * 128;

  // Stage q segment (128 x 64 halves = 1024 x 16B) into LDS.
  for (int c = tid; c < 128 * 8; c += 256) {
    const int r = c >> 3, cc = (c & 7) * 8;
    *(v8h*)&qsh[r][cc] = *(const v8h*)(qp + (size_t)(rb + r) * ADIM + cc);
  }

  // Overlap: load this wave's k strip A-fragments from global.
  v16h ak0 = a_frag_f16(kp, ADIM, rb + wv * 16 + lrow, 0,  half);
  v16h ak1 = a_frag_f16(kp, ADIM, rb + wv * 16 + lrow, 32, half);
  __syncthreads();

  // M = k_seg @ q_seg^T : contraction over A = 64 -> two K=32 WMMA steps.
  for (int c = 0; c < 8; ++c) {
    v8f acc = {};
    acc = wmma32(ak0, b_frag_f16(&qsh[0][0], 72, c * 16 + lrow, 0,  half), acc);
    acc = wmma32(ak1, b_frag_f16(&qsh[0][0], 72, c * 16 + lrow, 32, half), acc);
#pragma unroll
    for (int j = 0; j < 8; ++j)
      Msh[wv * 16 + j + 8 * half][c * 16 + lrow] = acc[j];
  }
  __syncthreads();

  // Row-wise softmax (axis=2). Threads 0..127 each own one row.
  if (tid < 128) {
    float mx = -3.4e38f;
    for (int m = 0; m < 128; ++m) mx = fmaxf(mx, Msh[tid][m]);
    float s = 0.0f;
    for (int m = 0; m < 128; ++m) { float e = __expf(Msh[tid][m] - mx); Msh[tid][m] = e; s += e; }
    const float inv = 1.0f / s;
    for (int m = 0; m < 128; ++m) Msh[tid][m] *= inv;
  }
  __syncthreads();

  // loss_vec = column sums (+EPS); loss += sum(-log(v)*v)
  if (tid < 128) {
    float s = 0.0f;
    for (int l = 0; l < 128; ++l) s += Msh[l][tid];
    const float v = s + EPSV;
    red[tid] = -__logf(v) * v;
  }
  __syncthreads();
  if (tid == 0) {
    float t = 0.0f;
    for (int i = 0; i < 128; ++i) t += red[i];
    atomicAdd(loss, t);
  }
  __syncthreads();

  // W = (attn + EPS) / L
  for (int idx = tid; idx < 128 * 128; idx += 256) {
    const int r = idx >> 7, m = idx & 127;
    Msh[r][m] = (Msh[r][m] + EPSV) * INVL;
  }
  __syncthreads();

  // Sinkhorn: alternate column-normalize (ax=1) / row-normalize (ax=2),
  // starting with columns; each followed by /L (folded as INVL/sum).
  for (int it = 0; it < 5; ++it) {
    if ((it & 1) == 0) {             // normalize columns
      if (tid < 128) {
        float s = 0.0f;
        for (int l = 0; l < 128; ++l) s += Msh[l][tid];
        red[tid] = INVL / s;
      }
      __syncthreads();
      for (int idx = tid; idx < 128 * 128; idx += 256)
        Msh[idx >> 7][idx & 127] *= red[idx & 127];
    } else {                         // normalize rows
      if (tid < 128) {
        float s = 0.0f;
        for (int m = 0; m < 128; ++m) s += Msh[tid][m];
        red[tid] = INVL / s;
      }
      __syncthreads();
      for (int idx = tid; idx < 128 * 128; idx += 256)
        Msh[idx >> 7][idx & 127] *= red[idx >> 7];
    }
    __syncthreads();
  }

  // Coalesced writeout of W for this segment.
  float* outp = Wout + (size_t)seg * (128 * 128);
  for (int idx = tid; idx < 128 * 128; idx += 256)
    outp[idx] = Msh[idx >> 7][idx & 127];
}

// ---------------------------------------------------------------------------
extern "C" void kernel_launch(void* const* d_in, const int* in_sizes, int n_in,
                              void* d_out, int out_size, void* d_ws, size_t ws_size,
                              hipStream_t stream) {
  (void)in_sizes; (void)n_in; (void)out_size; (void)ws_size;

  const float* yhat = (const float*)d_in[0];
  const float* y    = (const float*)d_in[1];
  const float* k0w  = (const float*)d_in[2];
  const float* k0b  = (const float*)d_in[3];
  const float* k1w  = (const float*)d_in[4];
  const float* k1b  = (const float*)d_in[5];
  const float* q0w  = (const float*)d_in[6];
  const float* q0b  = (const float*)d_in[7];
  const float* q1w  = (const float*)d_in[8];
  const float* q1b  = (const float*)d_in[9];

  _Float16* kp   = (_Float16*)d_ws;                      // 32 MB
  _Float16* qp   = kp + (size_t)NROWS * ADIM;            // 32 MB
  _Float16* wbuf = qp + (size_t)NROWS * ADIM;            // 320 KB f16 weights

  float* Wout = (float*)d_out;                           // NSEG*128*128 floats
  float* loss = Wout + (size_t)NSEG * SEGLEN * SEGLEN;   // final scalar

  hipMemsetAsync(loss, 0, sizeof(float), stream);        // graph-capturable

  cvt_weights_kernel<<<(WTOTAL + 255) / 256, 256, 0, stream>>>(k0w, k1w, q0w, q1w, wbuf);

  proj_kernel<<<NSEG, 256, 0, stream>>>(y,    wbuf + W0K_OFF, k0b,
                                        wbuf + W1K_OFF, k1b, kp);
  proj_kernel<<<NSEG, 256, 0, stream>>>(yhat, wbuf + W0Q_OFF, q0b,
                                        wbuf + W1Q_OFF, q1b, qp);
  attn_kernel<<<NSEG, 256, 0, stream>>>(kp, qp, Wout, loss);
}
